// Seq2Seq_87505663689391
// MI455X (gfx1250) — matrix operand order repaired and checked by reading, not compiled, not run on hardware
//
#include <hip/hip_runtime.h>
#include <cstddef>

// ---------------------------------------------------------------------------
// Seq2Seq (2-layer LSTM enc + 2-layer LSTM dec + FC) for MI455X / gfx1250.
// bf16 WMMA (v_wmma_f32_16x16x32_bf16) for all matmuls, fp32 cell state.
// GEMMs use 16x64 per-wave tiles: A fragment loaded once per K-tile and
// reused across 4 WMMAs (loads/WMMA ~2.5 instead of 4).
// ---------------------------------------------------------------------------

typedef __attribute__((ext_vector_type(16))) __bf16          v16bf;
typedef __attribute__((ext_vector_type(8)))  float           v8f;
typedef __attribute__((ext_vector_type(8)))  unsigned short  us8;
typedef __attribute__((ext_vector_type(16))) unsigned short  us16;

#define VSZ   32000
#define ESZ   512
#define HSZ   1024
#define BSZ   16
#define SSZ   128
#define GSZ   4096            // 4*H
#define T_ENC 128
#define T_DEC 127
#define BH    (BSZ*HSZ)       // 16384
#define BG    (BSZ*GSZ)       // 65536

// ---- fp32 -> bf16 (round to nearest even) ---------------------------------
__device__ __forceinline__ unsigned short f2bf(float f) {
  unsigned int u = __builtin_bit_cast(unsigned int, f);
  u += 0x7fffu + ((u >> 16) & 1u);
  return (unsigned short)(u >> 16);
}

// ---- WMMA fragment loads (16-bit operands, wave32, contiguous k order used in both A and B) -------------------
// A tile 16x32 (MxK), row-major source with leading dim ldA:
//   lane L: M = L%16, K = e + 16*(L>=16)  (two contiguous 16B loads)
__device__ __forceinline__ v16bf load_a_frag(const unsigned short* A, int ldA, int lane) {
  int l16 = lane & 15, hi = lane >> 4;
  const unsigned short* p = A + (size_t)l16 * ldA + hi * 16;
  us8 x0 = *(const us8*)(p);
  us8 x1 = *(const us8*)(p + 8);
  us16 c;
#pragma unroll
  for (int i = 0; i < 8; ++i) { c[i] = x0[i]; c[8 + i] = x1[i]; }
  return __builtin_bit_cast(v16bf, c);
}

// B tile 32x16 (KxN) taken from row-major W[N][K] (i.e. B = W^T chunk):
//   lane L: N = L%16, K = e + 16*(L>=16)  (one contiguous 32B load per lane)
__device__ __forceinline__ v16bf load_b_frag(const unsigned short* W, int ldB, int lane) {
  int l16 = lane & 15, hi = lane >> 4;
  us16 x = *(const us16*)(W + (size_t)l16 * ldB + hi * 16);
  return __builtin_bit_cast(v16bf, x);
}

__device__ __forceinline__ v8f wmma_bf16(v16bf a, v16bf b, v8f c) {
  return __builtin_amdgcn_wmma_f32_16x16x32_bf16(false, a, false, b, (short)0, c,
                                                 false, false);
}

__device__ __forceinline__ float sigmoidf_(float x) {
  return 1.0f / (1.0f + __expf(-x));
}

// ---- utility kernels ------------------------------------------------------
__global__ void cvt_f32_bf16_kernel(const float* __restrict__ in,
                                    unsigned short* __restrict__ outp, size_t n) {
  size_t i = (size_t)blockIdx.x * blockDim.x + threadIdx.x;
  size_t stride = (size_t)gridDim.x * blockDim.x;
  for (; i < n; i += stride) outp[i] = f2bf(in[i]);
}

__global__ void zero_f32_kernel(float* __restrict__ p, int n) {
  int i = blockIdx.x * blockDim.x + threadIdx.x;
  if (i < n) p[i] = 0.0f;
}

__global__ void zero_bf16_kernel(unsigned short* __restrict__ p, int n) {
  int i = blockIdx.x * blockDim.x + threadIdx.x;
  if (i < n) p[i] = 0;
}

__global__ void copy_f32_kernel(const float* __restrict__ a, float* __restrict__ b, int n) {
  int i = blockIdx.x * blockDim.x + threadIdx.x;
  if (i < n) b[i] = a[i];
}

// out[b][0][:] = 0   (first timestep of output tensor)
__global__ void zero_out_t0_kernel(float* __restrict__ outp) {
  int i = blockIdx.x * blockDim.x + threadIdx.x;       // 16*32000
  if (i >= BSZ * VSZ) return;
  int b = i / VSZ, v = i % VSZ;
  outp[(size_t)b * SSZ * VSZ + v] = 0.0f;
}

// X[t][b][e] = bf16(emb[idx[b][t]][e]);  idx row stride = SSZ
__global__ void embed_kernel(const int* __restrict__ idx, const float* __restrict__ emb,
                             unsigned short* __restrict__ outp, int T) {
  size_t tid = (size_t)blockIdx.x * blockDim.x + threadIdx.x;
  size_t total = (size_t)T * BSZ * ESZ;
  if (tid >= total) return;
  int e = tid % ESZ;
  int b = (tid / ESZ) % BSZ;
  int t = tid / ((size_t)ESZ * BSZ);
  int token = idx[b * SSZ + t];
  outp[tid] = f2bf(emb[(size_t)token * ESZ + e]);
}

// ---- Gx = X @ Wih^T + bias   (parallel over all timesteps) ----------------
// X: bf16 [M][K] (M = T*16 rows, t-major), W: bf16 [4096][K], out: f32 [M][4096]
// One wave -> 16x64 output tile (4 accumulators, A reused across 4 WMMAs).
__global__ void gates_x_gemm_kernel(const unsigned short* __restrict__ X,
                                    const unsigned short* __restrict__ W,
                                    const float* __restrict__ bias,
                                    float* __restrict__ outp, int M, int K) {
  int lane = threadIdx.x & 31;
  int wave = threadIdx.x >> 5;
  int wg   = blockIdx.x * (blockDim.x >> 5) + wave;
  const int NG = GSZ / 64;                 // 64 n-groups of 4 tiles
  int total = (M / 16) * NG;
  if (wg >= total) return;
  int mt = wg / NG, ng = wg % NG;
  int n0 = ng * 64;
  int l16 = lane & 15, hi = lane >> 4;

  v8f acc0 = {0.f, 0.f, 0.f, 0.f, 0.f, 0.f, 0.f, 0.f};
  v8f acc1 = acc0, acc2 = acc0, acc3 = acc0;
  const unsigned short* Abase = X + (size_t)mt * 16 * K;
  const unsigned short* B0 = W + (size_t)(n0 +  0) * K;
  const unsigned short* B1 = W + (size_t)(n0 + 16) * K;
  const unsigned short* B2 = W + (size_t)(n0 + 32) * K;
  const unsigned short* B3 = W + (size_t)(n0 + 48) * K;
  int KT = K >> 5;
  for (int kt = 0; kt < KT; ++kt) {
    v16bf a = load_a_frag(Abase + kt * 32, K, lane);
    acc0 = wmma_bf16(a, load_b_frag(B0 + kt * 32, K, lane), acc0);
    acc1 = wmma_bf16(a, load_b_frag(B1 + kt * 32, K, lane), acc1);
    acc2 = wmma_bf16(a, load_b_frag(B2 + kt * 32, K, lane), acc2);
    acc3 = wmma_bf16(a, load_b_frag(B3 + kt * 32, K, lane), acc3);
  }

  float* orow = outp + n0 + l16;
#pragma unroll
  for (int v = 0; v < 8; ++v) {
    int m = mt * 16 + v + 8 * hi;          // C/D layout: M = vgpr + 8*(lane>=16)
    float* o = orow + (size_t)m * GSZ;
    o[0]  = acc0[v] + bias[n0 + l16];
    o[16] = acc1[v] + bias[n0 + 16 + l16];
    o[32] = acc2[v] + bias[n0 + 32 + l16];
    o[48] = acc3[v] + bias[n0 + 48 + l16];
  }
}

// ---- one recurrent LSTM step ----------------------------------------------
// 64 waves (16 WG x 4 waves); wave w owns hidden units j in [16w,16w+16) and
// computes all four gates for them -> local cell update, no cross-wave comms.
// gx: f32 [16][4096] (this timestep), hprev: bf16 [16][1024],
// whh: bf16 [4096][1024], c: f32 [16][1024] (in/out), hout: bf16 [16][1024]
__global__ void lstm_step_kernel(const float* __restrict__ gx,
                                 const unsigned short* __restrict__ hprev,
                                 const unsigned short* __restrict__ whh,
                                 float* __restrict__ c,
                                 unsigned short* __restrict__ hout) {
  int lane = threadIdx.x & 31;
  int wave = threadIdx.x >> 5;
  int w = blockIdx.x * (blockDim.x >> 5) + wave;   // 0..63
  int j0 = w * 16;
  int l16 = lane & 15, hi = lane >> 4;

  v8f aI = {0.f,0.f,0.f,0.f,0.f,0.f,0.f,0.f};
  v8f aF = aI, aG = aI, aO = aI;

  for (int kt = 0; kt < HSZ / 32; ++kt) {
    v16bf a  = load_a_frag(hprev + kt * 32, HSZ, lane);
    v16bf bi = load_b_frag(whh + ((size_t)(0 * HSZ + j0)) * HSZ + kt * 32, HSZ, lane);
    v16bf bf = load_b_frag(whh + ((size_t)(1 * HSZ + j0)) * HSZ + kt * 32, HSZ, lane);
    v16bf bg = load_b_frag(whh + ((size_t)(2 * HSZ + j0)) * HSZ + kt * 32, HSZ, lane);
    v16bf bo = load_b_frag(whh + ((size_t)(3 * HSZ + j0)) * HSZ + kt * 32, HSZ, lane);
    aI = wmma_bf16(a, bi, aI);
    aF = wmma_bf16(a, bf, aF);
    aG = wmma_bf16(a, bg, aG);
    aO = wmma_bf16(a, bo, aO);
  }

  int j = j0 + l16;
#pragma unroll
  for (int v = 0; v < 8; ++v) {
    int b = v + 8 * hi;
    const float* gxb = gx + (size_t)b * GSZ;
    float i_ = sigmoidf_(aI[v] + gxb[j]);
    float f_ = sigmoidf_(aF[v] + gxb[j + HSZ]);
    float g_ = tanhf(aG[v] + gxb[j + 2 * HSZ]);
    float o_ = sigmoidf_(aO[v] + gxb[j + 3 * HSZ]);
    size_t ci = (size_t)b * HSZ + j;
    float cn = f_ * c[ci] + i_ * g_;
    c[ci] = cn;
    hout[ci] = f2bf(o_ * tanhf(cn));
  }
}

// ---- final FC: logits = D1 @ fc_W^T + fc_b, scattered to out[b][t+1][v] ----
// X: bf16 [2032][1024] (row m = t*16+b), W: bf16 [32000][1024]
// One wave -> 16x64 output tile (A fragment reused across 4 WMMAs).
__global__ void fc_gemm_kernel(const unsigned short* __restrict__ X,
                               const unsigned short* __restrict__ W,
                               const float* __restrict__ bias,
                               float* __restrict__ outp) {
  int lane = threadIdx.x & 31;
  int wave = threadIdx.x >> 5;
  int wg   = blockIdx.x * (blockDim.x >> 5) + wave;
  const int NG = VSZ / 64;                 // 500 n-groups
  int total = T_DEC * NG;                  // 63500 waves
  if (wg >= total) return;
  int mt = wg / NG, ng = wg % NG;          // mt == timestep t
  int n0 = ng * 64;
  int l16 = lane & 15, hi = lane >> 4;

  v8f acc0 = {0.f, 0.f, 0.f, 0.f, 0.f, 0.f, 0.f, 0.f};
  v8f acc1 = acc0, acc2 = acc0, acc3 = acc0;
  const unsigned short* Abase = X + (size_t)mt * 16 * HSZ;
  const unsigned short* B0 = W + (size_t)(n0 +  0) * HSZ;
  const unsigned short* B1 = W + (size_t)(n0 + 16) * HSZ;
  const unsigned short* B2 = W + (size_t)(n0 + 32) * HSZ;
  const unsigned short* B3 = W + (size_t)(n0 + 48) * HSZ;
  for (int kt = 0; kt < HSZ / 32; ++kt) {
    v16bf a = load_a_frag(Abase + kt * 32, HSZ, lane);
    acc0 = wmma_bf16(a, load_b_frag(B0 + kt * 32, HSZ, lane), acc0);
    acc1 = wmma_bf16(a, load_b_frag(B1 + kt * 32, HSZ, lane), acc1);
    acc2 = wmma_bf16(a, load_b_frag(B2 + kt * 32, HSZ, lane), acc2);
    acc3 = wmma_bf16(a, load_b_frag(B3 + kt * 32, HSZ, lane), acc3);
  }

#pragma unroll
  for (int v = 0; v < 8; ++v) {
    int b = v + 8 * hi;
    float* o = outp + ((size_t)b * SSZ + mt + 1) * VSZ + n0 + l16;
    o[0]  = acc0[v] + bias[n0 + l16];
    o[16] = acc1[v] + bias[n0 + 16 + l16];
    o[32] = acc2[v] + bias[n0 + 32 + l16];
    o[48] = acc3[v] + bias[n0 + 48 + l16];
  }
}

// ---------------------------------------------------------------------------
extern "C" void kernel_launch(void* const* d_in, const int* in_sizes, int n_in,
                              void* d_out, int out_size, void* d_ws, size_t ws_size,
                              hipStream_t stream) {
  const int*   src      = (const int*)  d_in[0];
  const int*   trg      = (const int*)  d_in[1];
  const float* enc_emb  = (const float*)d_in[2];
  const float* dec_emb  = (const float*)d_in[3];
  const float* enc_Wih0 = (const float*)d_in[4];
  const float* enc_Whh0 = (const float*)d_in[5];
  const float* enc_b0   = (const float*)d_in[6];
  const float* enc_Wih1 = (const float*)d_in[7];
  const float* enc_Whh1 = (const float*)d_in[8];
  const float* enc_b1   = (const float*)d_in[9];
  const float* dec_Wih0 = (const float*)d_in[10];
  const float* dec_Whh0 = (const float*)d_in[11];
  const float* dec_b0   = (const float*)d_in[12];
  const float* dec_Wih1 = (const float*)d_in[13];
  const float* dec_Whh1 = (const float*)d_in[14];
  const float* dec_b1   = (const float*)d_in[15];
  const float* fc_W     = (const float*)d_in[16];
  const float* fc_b     = (const float*)d_in[17];
  float* outp = (float*)d_out;

  // --- workspace bump allocator ---
  char* base = (char*)d_ws;
  size_t off = 0;
  auto alloc = [&](size_t bytes) -> void* {
    void* p = base + off;
    off += (bytes + 255) & ~(size_t)255;
    return p;
  };
  auto allocBF = [&](size_t nelem) { return (unsigned short*)alloc(nelem * 2); };
  auto allocF  = [&](size_t nelem) { return (float*)alloc(nelem * 4); };

  unsigned short* eWih0 = allocBF((size_t)GSZ * ESZ);
  unsigned short* eWhh0 = allocBF((size_t)GSZ * HSZ);
  unsigned short* eWih1 = allocBF((size_t)GSZ * HSZ);
  unsigned short* eWhh1 = allocBF((size_t)GSZ * HSZ);
  unsigned short* dWih0 = allocBF((size_t)GSZ * ESZ);
  unsigned short* dWhh0 = allocBF((size_t)GSZ * HSZ);
  unsigned short* dWih1 = allocBF((size_t)GSZ * HSZ);
  unsigned short* dWhh1 = allocBF((size_t)GSZ * HSZ);
  unsigned short* fcWbf = allocBF((size_t)VSZ * HSZ);
  unsigned short* Xe    = allocBF((size_t)T_ENC * BSZ * ESZ);
  unsigned short* Xd    = allocBF((size_t)T_DEC * BSZ * ESZ);
  unsigned short* Ye0   = allocBF((size_t)T_ENC * BH);
  unsigned short* Ye1   = allocBF((size_t)T_ENC * BH);
  unsigned short* Yd0   = allocBF((size_t)T_DEC * BH);
  unsigned short* Yd1   = allocBF((size_t)T_DEC * BH);
  unsigned short* h0z   = allocBF(BH);
  float* Gx  = allocF((size_t)T_ENC * BG);
  float* cE0 = allocF(BH);
  float* cE1 = allocF(BH);
  float* cD0 = allocF(BH);
  float* cD1 = allocF(BH);

  // --- weight conversion fp32 -> bf16 (deterministic, every call) ---
  auto conv = [&](const float* s, unsigned short* d, size_t n) {
    cvt_f32_bf16_kernel<<<2048, 256, 0, stream>>>(s, d, n);
  };
  conv(enc_Wih0, eWih0, (size_t)GSZ * ESZ);
  conv(enc_Whh0, eWhh0, (size_t)GSZ * HSZ);
  conv(enc_Wih1, eWih1, (size_t)GSZ * HSZ);
  conv(enc_Whh1, eWhh1, (size_t)GSZ * HSZ);
  conv(dec_Wih0, dWih0, (size_t)GSZ * ESZ);
  conv(dec_Whh0, dWhh0, (size_t)GSZ * HSZ);
  conv(dec_Wih1, dWih1, (size_t)GSZ * HSZ);
  conv(dec_Whh1, dWhh1, (size_t)GSZ * HSZ);
  conv(fc_W,     fcWbf, (size_t)VSZ * HSZ);

  // --- init states / output t=0 ---
  zero_bf16_kernel<<<(BH + 255) / 256, 256, 0, stream>>>(h0z, BH);
  zero_f32_kernel<<<(BH + 255) / 256, 256, 0, stream>>>(cE0, BH);
  zero_f32_kernel<<<(BH + 255) / 256, 256, 0, stream>>>(cE1, BH);
  zero_out_t0_kernel<<<(BSZ * VSZ + 255) / 256, 256, 0, stream>>>(outp);

  // --- embeddings (direct to bf16, time-major [T][B][E]) ---
  {
    size_t ne = (size_t)T_ENC * BSZ * ESZ;
    embed_kernel<<<(unsigned)((ne + 255) / 256), 256, 0, stream>>>(src, enc_emb, Xe, T_ENC);
    size_t nd = (size_t)T_DEC * BSZ * ESZ;
    embed_kernel<<<(unsigned)((nd + 255) / 256), 256, 0, stream>>>(trg, dec_emb, Xd, T_DEC);
  }

  // --- one LSTM layer pass: hoisted x-projection GEMM, then recurrence ---
  auto run_layer = [&](const unsigned short* X, int K, int T,
                       const unsigned short* Wih, const float* bias,
                       const unsigned short* Whh, float* cbuf,
                       const unsigned short* h0, unsigned short* Y) {
    int M = T * BSZ;
    int waves = (M / 16) * (GSZ / 64);
    gates_x_gemm_kernel<<<(waves + 7) / 8, 256, 0, stream>>>(X, Wih, bias, Gx, M, K);
    for (int t = 0; t < T; ++t) {
      const unsigned short* hp = (t == 0) ? h0 : (Y + (size_t)(t - 1) * BH);
      lstm_step_kernel<<<16, 128, 0, stream>>>(Gx + (size_t)t * BG, hp, Whh, cbuf,
                                               Y + (size_t)t * BH);
    }
  };

  // encoder
  run_layer(Xe,  ESZ, T_ENC, eWih0, enc_b0, eWhh0, cE0, h0z, Ye0);
  run_layer(Ye0, HSZ, T_ENC, eWih1, enc_b1, eWhh1, cE1, h0z, Ye1);

  // decoder init states = encoder finals
  copy_f32_kernel<<<(BH + 255) / 256, 256, 0, stream>>>(cE0, cD0, BH);
  copy_f32_kernel<<<(BH + 255) / 256, 256, 0, stream>>>(cE1, cD1, BH);
  const unsigned short* hEnc0 = Ye0 + (size_t)(T_ENC - 1) * BH;
  const unsigned short* hEnc1 = Ye1 + (size_t)(T_ENC - 1) * BH;

  // decoder
  run_layer(Xd,  ESZ, T_DEC, dWih0, dec_b0, dWhh0, cD0, hEnc0, Yd0);
  run_layer(Yd0, HSZ, T_DEC, dWih1, dec_b1, dWhh1, cD1, hEnc1, Yd1);

  // final projection to vocab
  {
    int waves = T_DEC * (VSZ / 64);        // 63500
    fc_gemm_kernel<<<(waves + 7) / 8, 256, 0, stream>>>(Yd1, fcWbf, fc_b, outp);
  }

  (void)in_sizes; (void)n_in; (void)out_size; (void)ws_size;
}
